// classSR_3class_fsrcnn_net_51771535786326
// MI455X (gfx1250) — compile-verified, matrix operand order checked by reading, not run on hardware
//
#include <hip/hip_runtime.h>
#include <hip/hip_bf16.h>

// ---------------------------------------------------------------------------
// ClassSR (3-class FSRCNN) for MI455X / gfx1250, WMMA implicit-GEMM pipeline.
//
// All activations live in NHWC with channels padded to a multiple of 8
// (Cp), im2col K-order is (tap, channel): k = (ky*KW+kx)*Cp + ic.  With this
// layout each 8-wide K segment of a v_wmma_f32_16x16x32_f16 A-fragment is a
// contiguous 16B run -> one global_load_b128, branchless validity select.
// B fragments come from N-major packed f16 weights (two b128s / fragment).
// The stride-4 9x9 transposed conv is one GEMM with N = 48 (16 sub-pixel
// phases x 3 channels), scattering f32 directly into d_out.  Expert pipelines
// skip images the classifier routed elsewhere (output-identical to reference).
// ---------------------------------------------------------------------------

typedef __attribute__((ext_vector_type(16))) _Float16 v16h;
typedef __attribute__((ext_vector_type(8)))  float    v8f;
typedef _Float16 h16;

union FragH { v16h v; unsigned int u[8]; };

struct GemmDesc {
  const h16*   A;      // activations [img][H][W][Cp], f16
  const h16*   Wt;     // packed weights [Npad][Kpad], f16 (N-major)
  const float* bias;   // [Cout] or null
  const float* alpha;  // [Cout] PReLU slopes (act==1)
  h16*         Out;    // [img][OH][OW][OCp], f16
  const int*   sel;    // per-image routing (null = no routing)
  int          expert;
  int H, W, pad, stride;
  int Cout, OCp, OH, OW;
  int Kreal, Kpad;     // Kreal = KH*KW*Cp (mult of 8), Kpad mult of 32
  int act;             // 0 none, 1 prelu(alpha), 2 leaky(lrelu)
  float lrelu;
};

// One wave -> one 16x16 f32 tile of A(MxK) * W^T(KxN).
// A fragment (ISA 7.12.2): lane = M row, VGPRs 0..3 hold K = kt+8h..+7,
// VGPRs 4..7 hold K = kt+16+8h..+7  ->  two b128 loads from NHWC memory.
template<int KH, int KW, int CP>
__device__ __forceinline__ v8f gemm_tile_wmma(const GemmDesc& g, int m0, int img, int n0)
{
  const int lane = threadIdx.x;
  const int posPerImg = g.OH * g.OW;
  int pos = (m0 - img * posPerImg) + (lane & 15);
  int oy  = pos / g.OW;
  int ox  = pos - oy * g.OW;
  int iy0 = oy * g.stride - g.pad;
  int ix0 = ox * g.stride - g.pad;
  const h16* Abase = g.A + (size_t)img * ((size_t)g.H * g.W * CP);
  const int  khalf = (lane >> 4) << 3;
  const h16* Wrow  = g.Wt + (size_t)(n0 + (lane & 15)) * g.Kpad;

  v8f acc = {};
  for (int kt = 0; kt < g.Kpad; kt += 32) {
    FragH a, b;
    // B: contiguous in packed weights
    *(uint4*)&b.u[0] = *(const uint4*)(Wrow + kt + khalf);
    *(uint4*)&b.u[4] = *(const uint4*)(Wrow + kt + khalf + 16);
    // A: two 8-element segments, each inside one (ky,kx) tap
#pragma unroll
    for (int s = 0; s < 2; ++s) {
      int k   = kt + khalf + 16 * s;
      int tap = k / CP;                      // compile-time divisor
      int ic  = k - tap * CP;
      int ky  = tap / KW;
      int kx  = tap - ky * KW;
      int iy = iy0 + ky, ix = ix0 + kx;
      bool valid = (k < g.Kreal) &
                   ((unsigned)iy < (unsigned)g.H) &
                   ((unsigned)ix < (unsigned)g.W);
      int off32 = (iy * g.W + ix) * CP + ic;
      size_t offs = valid ? (size_t)off32 : (size_t)0;
      uint4 q = *(const uint4*)(Abase + offs);   // 16B aligned (CP % 8 == 0)
      q.x = valid ? q.x : 0u;
      q.y = valid ? q.y : 0u;
      q.z = valid ? q.z : 0u;
      q.w = valid ? q.w : 0u;
      *(uint4*)&a.u[4 * s] = q;
    }
    acc = __builtin_amdgcn_wmma_f32_16x16x32_f16(false, a.v, false, b.v,
                                                 (short)0, acc, false, false);
  }
  return acc;
}

// Generic conv-as-GEMM, fused bias + PReLU / leaky-ReLU, NHWC f16 output
// (pad channels written as zero so downstream GEMMs stay exact).
template<int KH, int KW, int CP>
__global__ __launch_bounds__(32) void conv_gemm(GemmDesc g)
{
  const int lane = threadIdx.x;
  const int m0 = blockIdx.x << 4;
  const int posPerImg = g.OH * g.OW;       // 64 or 1024: multiple of 16
  const int img = m0 / posPerImg;
  if (g.sel && g.sel[img] != g.expert) return;
  const int n0 = blockIdx.y << 4;

  v8f acc = gemm_tile_wmma<KH, KW, CP>(g, m0, img, n0);

  int n = n0 + (lane & 15);                // C/D layout: lane = N column
  if (n >= g.OCp) return;
  bool real = (n < g.Cout);
  float bs = (real && g.bias) ? g.bias[n] : 0.f;
  float al = (g.act == 1 && real) ? g.alpha[n] : g.lrelu;
  int  mbase = (m0 - img * posPerImg) + ((lane >> 4) << 3);
  h16* Obase = g.Out + ((size_t)img * posPerImg + mbase) * g.OCp + n;
#pragma unroll
  for (int v = 0; v < 8; ++v) {            // VGPR v = M row v + 8*(lane>>4)
    float val = acc[v] + bs;
    if (g.act) val = (val > 0.f) ? val : al * val;
    if (!real) val = 0.f;
    Obase[(size_t)v * g.OCp] = (h16)val;
  }
}

// Transposed conv (stride 4, 9x9) as GEMM over a 3x3 neighborhood of the
// expand output: K = 9*Cp, N = 48 = (py*4+px)*3 + oc.  f32 scatter to d_out.
template<int CP>
__global__ __launch_bounds__(32) void deconv_gemm(GemmDesc g, float* out)
{
  const int lane = threadIdx.x;
  const int m0 = blockIdx.x << 4;
  const int img = m0 >> 10;                // 1024 base positions / image
  if (g.sel[img] != g.expert) return;
  const int n0 = blockIdx.y << 4;

  v8f acc = gemm_tile_wmma<3, 3, CP>(g, m0, img, n0);

  int n  = n0 + (lane & 15);               // 0..47
  int p  = n / 3;
  int oc = n - p * 3;
  int py = p >> 2, px = p & 3;
  float bs = g.bias[oc];
  int    tbase = (m0 & 1023) + ((lane >> 4) << 3);
  float* Obase = out + ((size_t)img * 3 + oc) * 16384;
#pragma unroll
  for (int v = 0; v < 8; ++v) {
    int t = tbase + v;
    int ty = t >> 5, tx = t & 31;
    Obase[(size_t)((ty << 2) + py) * 128 + ((tx << 2) + px)] = acc[v] + bs;
  }
}

// ------------------------------- helpers -----------------------------------

// x (NCHW f32, 3ch) -> NHWC f16 [img][32][32][8], channels 3..7 = 0
__global__ void cvt_x_nhwc(const float* x, h16* dst, int n)
{
  int i = blockIdx.x * blockDim.x + threadIdx.x;
  if (i >= n) return;
  int c   = i & 7;
  int pos = (i >> 3) & 1023;
  int img = i >> 13;
  dst[i] = (h16)((c < 3) ? x[((size_t)img * 3 + c) * 1024 + pos] : 0.f);
}

// OIHW f32 -> [Npad][Kpad] f16 with K-order k = tap*Cp + ic
__global__ void pack_w(const float* src, h16* dst, int OC, int Cin, int KHW,
                       int Cp, int Npad, int Kpad)
{
  int i = blockIdx.x * blockDim.x + threadIdx.x;
  if (i >= Npad * Kpad) return;
  int n = i / Kpad, k = i - n * Kpad;
  int tap = k / Cp, ic = k - tap * Cp;
  float v = 0.f;
  if (n < OC && tap < KHW && ic < Cin)
    v = src[((size_t)n * Cin + ic) * KHW + tap];
  dst[i] = (h16)v;
}

// dw (d,3,9,9) -> sparse-expanded deconv weights [48][Kpad], K = tap*Cp + ic.
// y[oc,4t+py,4u+px] = sum_{ic,dy,dx} h[ic,t+dy,u+dx] * wt[oc,ic,ky,kx],
// ky = 5-py+4*dy, kx = 5-px+4*dx, wt = flip(dw).transpose(1,0,2,3).
__global__ void pack_deconv(const float* dw, h16* dst, int dch, int Cp, int Kpad)
{
  int i = blockIdx.x * blockDim.x + threadIdx.x;
  if (i >= 48 * Kpad) return;
  int n = i / Kpad, k = i - n * Kpad;
  int p = n / 3, oc = n - p * 3;
  int py = p >> 2, px = p & 3;
  int tap = k / Cp, ic = k - tap * Cp;
  float v = 0.f;
  if (tap < 9 && ic < dch) {
    int dy = tap / 3 - 1, dx = (tap - (tap / 3) * 3) - 1;
    int ky = 5 - py + 4 * dy, kx = 5 - px + 4 * dx;
    if (ky >= 0 && ky < 9 && kx >= 0 && kx < 9)
      v = dw[(((size_t)ic * 3 + oc) * 9 + (8 - ky)) * 9 + (8 - kx)];
  }
  dst[i] = (h16)v;
}

__global__ void zero_counts(float* p) { if (threadIdx.x < 3) p[threadIdx.x] = 0.f; }

// Per image: mean-pool c4 (NHWC [8][8][32]), 32->3 linear, argmax, count.
__global__ __launch_bounds__(32) void cls_tail(const h16* c4, const float* lw,
                                               const float* lb, int* idx, float* counts)
{
  int img = blockIdx.x, lane = threadIdx.x;
  const h16* base = c4 + (size_t)img * 2048 + lane;
  float mean = 0.f;
  for (int i = 0; i < 64; ++i) mean += (float)base[i * 32];
  mean *= (1.f / 64.f);
  float best = -3.4e38f; int bi = 0;
  for (int j = 0; j < 3; ++j) {
    float part = lw[j * 32 + lane] * mean;
    for (int off = 16; off; off >>= 1) part += __shfl_xor(part, off);
    float logit = part + lb[j];
    if (logit > best) { best = logit; bi = j; }   // first-max tie break
  }
  if (lane == 0) { idx[img] = bi; atomicAdd(&counts[bi], 1.f); }
}

// ------------------------------ host side ----------------------------------

static void launch_conv(hipStream_t st, const GemmDesc& g, int KH, int CP,
                        int nImg, int Npad)
{
  dim3 grid((unsigned)(nImg * g.OH * g.OW / 16), (unsigned)(Npad / 16));
  if (KH == 4)      conv_gemm<4, 4, 8><<<grid, 32, 0, st>>>(g);
  else if (KH == 5) conv_gemm<5, 5, 8><<<grid, 32, 0, st>>>(g);
  else if (KH == 3) {
    if (CP == 16)      conv_gemm<3, 3, 16><<<grid, 32, 0, st>>>(g);
    else if (CP == 40) conv_gemm<3, 3, 40><<<grid, 32, 0, st>>>(g);
    else               conv_gemm<3, 3, 56><<<grid, 32, 0, st>>>(g);
  } else {
    if (CP == 16)      conv_gemm<1, 1, 16><<<grid, 32, 0, st>>>(g);
    else if (CP == 40) conv_gemm<1, 1, 40><<<grid, 32, 0, st>>>(g);
    else if (CP == 56) conv_gemm<1, 1, 56><<<grid, 32, 0, st>>>(g);
    else               conv_gemm<1, 1, 128><<<grid, 32, 0, st>>>(g);
  }
}

static GemmDesc mkd(const h16* A, const h16* W, const float* bias, const float* alpha,
                    int act, float lrelu, h16* Out,
                    int H, int Wi, int pad, int stride,
                    int Cout, int OCp, int OH, int OW, int Kreal, int Kpad,
                    const int* sel, int expert)
{
  GemmDesc g;
  g.A = A; g.Wt = W; g.bias = bias; g.alpha = alpha; g.Out = Out;
  g.sel = sel; g.expert = expert;
  g.H = H; g.W = Wi; g.pad = pad; g.stride = stride;
  g.Cout = Cout; g.OCp = OCp; g.OH = OH; g.OW = OW;
  g.Kreal = Kreal; g.Kpad = Kpad; g.act = act; g.lrelu = lrelu;
  return g;
}

extern "C" void kernel_launch(void* const* d_in, const int* in_sizes, int n_in,
                              void* d_out, int out_size, void* d_ws, size_t ws_size,
                              hipStream_t stream)
{
  (void)n_in; (void)out_size; (void)ws_size;
  const int    NIMG = 1024;
  const size_t OUTN = (size_t)NIMG * 3 * 128 * 128;

  // ---- input mapping (detect flatten order via x's unique size) ----
  auto F = [&](int i) { return (const float*)d_in[i]; };
  const float *X, *c0w, *c0b, *c1w, *c1b, *c2w, *c2b, *c3w, *c3b, *c4w, *c4b, *lw, *lb;
  struct Exp { const float *hw,*hb,*ha,*sw,*sb,*sa,*mw[4],*mb[4],*ma[4],*ew,*eb,*ea,*dwt,*db; int d; } ex[3];

  bool orderB = (in_sizes[0] == 1024 * 3 * 32 * 32);   // insertion order, x first
  if (orderB) {
    X = F(0);
    c0w=F(1);  c0b=F(2);  c1w=F(3);  c1b=F(4);  c2w=F(5);  c2b=F(6);
    c3w=F(7);  c3b=F(8);  c4w=F(9);  c4b=F(10); lw=F(11);  lb=F(12);
    for (int e = 0; e < 3; ++e) {
      int s = 13 + 23 * e;
      ex[e].hw=F(s+0); ex[e].hb=F(s+1); ex[e].ha=F(s+2);
      ex[e].sw=F(s+3); ex[e].sb=F(s+4); ex[e].sa=F(s+5);
      for (int i = 0; i < 4; ++i) { ex[e].mw[i]=F(s+6+i); ex[e].mb[i]=F(s+10+i); ex[e].ma[i]=F(s+14+i); }
      ex[e].ew=F(s+18); ex[e].eb=F(s+19); ex[e].ea=F(s+20);
      ex[e].dwt=F(s+21); ex[e].db=F(s+22);
    }
  } else {                                              // jax tree order (alphabetical), x last
    c0b=F(0);  c0w=F(1);  c1b=F(2);  c1w=F(3);  c2b=F(4);  c2w=F(5);
    c3b=F(6);  c3w=F(7);  c4b=F(8);  c4w=F(9);  lb=F(10);  lw=F(11);
    for (int e = 0; e < 3; ++e) {
      int s = 12 + 23 * e;
      ex[e].db=F(s+0);  ex[e].dwt=F(s+1);
      ex[e].ea=F(s+2);  ex[e].eb=F(s+3);  ex[e].ew=F(s+4);
      ex[e].ha=F(s+5);  ex[e].hb=F(s+6);  ex[e].hw=F(s+7);
      for (int i = 0; i < 4; ++i) { ex[e].ma[i]=F(s+8+i); ex[e].mb[i]=F(s+12+i); ex[e].mw[i]=F(s+16+i); }
      ex[e].sa=F(s+20); ex[e].sb=F(s+21); ex[e].sw=F(s+22);
    }
    X = F(81);
  }
  ex[0].d = 16; ex[1].d = 36; ex[2].d = 56;

  // ---- workspace layout ----
  char* wsp = (char*)d_ws;
  size_t off = 0;
  auto alloc = [&](size_t bytes) -> void* {
    void* p = wsp + off; off += (bytes + 255) & ~(size_t)255; return p;
  };
  h16* xh   = (h16*)alloc((size_t)1024 * 1024 * 8 * 2);    // x NHWC Cp=8
  h16* bufA = (h16*)alloc((size_t)1024 * 1024 * 64 * 2);   // head/expand out, cls ping
  h16* bufB = (h16*)alloc((size_t)1024 * 1024 * 16 * 2 + (size_t)1024*64*128*2);
  h16* bufC = (h16*)alloc((size_t)1024 * 1024 * 16 * 2 + (size_t)1024*64*128*2);
  h16* wc0  = (h16*)alloc(128 * 128 * 2);
  h16* wc1  = (h16*)alloc(128 * 128 * 2);
  h16* wc2  = (h16*)alloc(128 * 128 * 2);
  h16* wc3  = (h16*)alloc(128 * 128 * 2);
  h16* wc4  = (h16*)alloc(32 * 128 * 2);
  h16* wh   = (h16*)alloc(64 * 224 * 2);
  h16* wsr  = (h16*)alloc(16 * 64 * 2);
  h16* wm[4];
  for (int i = 0; i < 4; ++i) wm[i] = (h16*)alloc(16 * 160 * 2);
  h16* wex  = (h16*)alloc(64 * 32 * 2);
  h16* wdc  = (h16*)alloc(48 * 512 * 2);
  int* idx  = (int*)alloc(1024 * 4);

  float* out    = (float*)d_out;
  float* counts = out + OUTN;

#define PACK(src, dst, OC, CIN, KHW, CP, NP, KP) \
  pack_w<<<((NP) * (KP) + 255) / 256, 256, 0, stream>>>(src, dst, OC, CIN, KHW, CP, NP, KP)

  // ---- prologue ----
  cvt_x_nhwc<<<(1024 * 1024 * 8 + 255) / 256, 256, 0, stream>>>(X, xh, 1024 * 1024 * 8);
  zero_counts<<<1, 32, 0, stream>>>(counts);

  // ---- classifier: 5 GEMMs + tail ----
  PACK(c0w, wc0, 128, 3,   16, 8,   128, 128);
  PACK(c1w, wc1, 128, 128, 1,  128, 128, 128);
  PACK(c2w, wc2, 128, 128, 1,  128, 128, 128);
  PACK(c3w, wc3, 128, 128, 1,  128, 128, 128);
  PACK(c4w, wc4, 32,  128, 1,  128, 32,  128);

  launch_conv(stream, mkd(xh,   wc0, c0b, nullptr, 2, 0.1f, bufA, 32,32, 0,4, 128,128, 8,8, 128,128, nullptr,0), 4, 8,   NIMG, 128);
  launch_conv(stream, mkd(bufA, wc1, c1b, nullptr, 2, 0.1f, bufB,  8, 8, 0,1, 128,128, 8,8, 128,128, nullptr,0), 1, 128, NIMG, 128);
  launch_conv(stream, mkd(bufB, wc2, c2b, nullptr, 2, 0.1f, bufA,  8, 8, 0,1, 128,128, 8,8, 128,128, nullptr,0), 1, 128, NIMG, 128);
  launch_conv(stream, mkd(bufA, wc3, c3b, nullptr, 2, 0.1f, bufB,  8, 8, 0,1, 128,128, 8,8, 128,128, nullptr,0), 1, 128, NIMG, 128);
  launch_conv(stream, mkd(bufB, wc4, c4b, nullptr, 0, 0.0f, bufC,  8, 8, 0,1, 32, 32,  8,8, 128,128, nullptr,0), 1, 128, NIMG, 32);
  cls_tail<<<NIMG, 32, 0, stream>>>(bufC, lw, lb, idx, counts);

  // ---- routed FSRCNN experts ----
  for (int e = 0; e < 3; ++e) {
    int d    = ex[e].d;
    int Cp   = (d + 7) & ~7;        // 16 / 40 / 56
    int dp16 = (d + 15) & ~15;      // 16 / 48 / 64
    int Kshr = (Cp + 31) & ~31;     // 32 / 64 / 64
    int Kdec = (9 * Cp + 31) & ~31; // 160 / 384 / 512

    PACK(ex[e].hw, wh, d, 3, 25, 8, dp16, 224);
    PACK(ex[e].sw, wsr, 12, d, 1, Cp, 16, Kshr);
    for (int i = 0; i < 4; ++i) PACK(ex[e].mw[i], wm[i], 12, 12, 9, 16, 16, 160);
    PACK(ex[e].ew, wex, d, 12, 1, 16, dp16, 32);
    pack_deconv<<<(48 * Kdec + 255) / 256, 256, 0, stream>>>(ex[e].dwt, wdc, d, Cp, Kdec);

    // head 5x5 pad2 (3 -> d), PReLU
    launch_conv(stream, mkd(xh, wh, ex[e].hb, ex[e].ha, 1, 0.f, bufA,
                            32,32, 2,1, d, Cp, 32,32, 200, 224, idx, e), 5, 8, NIMG, dp16);
    // shrink 1x1 (d -> 12), PReLU
    launch_conv(stream, mkd(bufA, wsr, ex[e].sb, ex[e].sa, 1, 0.f, bufB,
                            32,32, 0,1, 12, 16, 32,32, Cp, Kshr, idx, e), 1, Cp, NIMG, 16);
    // 4x mid 3x3 pad1 (12 -> 12), PReLU, ping-pong B<->C
    h16* pin = bufB; h16* pout = bufC;
    for (int i = 0; i < 4; ++i) {
      launch_conv(stream, mkd(pin, wm[i], ex[e].mb[i], ex[e].ma[i], 1, 0.f, pout,
                              32,32, 1,1, 12, 16, 32,32, 144, 160, idx, e), 3, 16, NIMG, 16);
      h16* t = pin; pin = pout; pout = t;
    }
    // expand 1x1 (12 -> d), PReLU  (pin == bufB after 4 swaps)
    launch_conv(stream, mkd(pin, wex, ex[e].eb, ex[e].ea, 1, 0.f, bufA,
                            32,32, 0,1, d, Cp, 32,32, 16, 32, idx, e), 1, 16, NIMG, dp16);
    // deconv as GEMM: K = 9*Cp, N = 48 phases*channels, f32 scatter into d_out
    GemmDesc gd = mkd(bufA, wdc, ex[e].db, nullptr, 0, 0.f, nullptr,
                      32,32, 1,1, 48, 48, 32,32, 9 * Cp, Kdec, idx, e);
    dim3 dgrid(65536, 3);
    if (Cp == 16)      deconv_gemm<16><<<dgrid, 32, 0, stream>>>(gd, out);
    else if (Cp == 40) deconv_gemm<40><<<dgrid, 32, 0, stream>>>(gd, out);
    else               deconv_gemm<56><<<dgrid, 32, 0, stream>>>(gd, out);
  }
#undef PACK
}
